// E_GCL_2774548873773
// MI455X (gfx1250) — matrix operand order repaired and checked by reading, not compiled, not run on hardware
//
#include <hip/hip_runtime.h>
#include <hip/hip_bf16.h>

// ---------- types ----------
typedef __bf16 bf16;
typedef bf16  v8bf  __attribute__((ext_vector_type(8)));
typedef bf16  v16bf __attribute__((ext_vector_type(16)));
typedef float v8f   __attribute__((ext_vector_type(8)));

#define C_DIM 128
#define H_DIM 128
#define EA_DIM 8
#define KP1 288          // 2C+1+EA = 265 padded to multiple of 32
#define KN1 256          // C + H for node MLP

union AFrag { v16bf v; v8bf h[2]; };

// =====================================================================
// prep: convert weights f32 -> bf16 column-major (K contiguous, padded),
// and zero the aggregation buffer (must happen every launch).
// =====================================================================
__global__ __launch_bounds__(256) void prep_kernel(
    const float* __restrict__ w_e1, const float* __restrict__ w_e2,
    const float* __restrict__ w_n1, const float* __restrict__ w_n2,
    bf16* __restrict__ w1, bf16* __restrict__ w2,
    bf16* __restrict__ wn1, bf16* __restrict__ wn2,
    float* __restrict__ agg, int aggN)
{
    int idx = blockIdx.x * blockDim.x + threadIdx.x;
    int stride = gridDim.x * blockDim.x;
    for (int i = idx; i < aggN; i += stride) agg[i] = 0.0f;
    for (int i = idx; i < KP1 * H_DIM; i += stride) {
        int n = i / KP1, k = i % KP1;
        w1[i] = (bf16)((k < 265) ? w_e1[k * H_DIM + n] : 0.0f);
    }
    for (int i = idx; i < H_DIM * H_DIM; i += stride) {
        int n = i >> 7, k = i & 127;
        w2[i] = (bf16)w_e2[k * H_DIM + n];
    }
    for (int i = idx; i < KN1 * H_DIM; i += stride) {
        int n = i >> 8, k = i & 255;
        wn1[i] = (bf16)w_n1[k * H_DIM + n];
    }
    for (int i = idx; i < H_DIM * C_DIM; i += stride) {
        int n = i >> 7, k = i & 127;
        wn2[i] = (bf16)w_n2[k * C_DIM + n];
    }
}

// =====================================================================
// edge kernel: 64 edges per block. Gather -> (265->128 ReLU) ->
// (128->128 ReLU) -> atomic scatter-add into agg. All GEMMs via
// v_wmma_f32_16x16x32_bf16; 8 waves, each owns a 16x64 output tile.
// =====================================================================
__global__ __launch_bounds__(256) void edge_kernel(
    const float* __restrict__ x, const float* __restrict__ coord,
    const int* __restrict__ ei, const float* __restrict__ ea,
    const bf16* __restrict__ w1, const float* __restrict__ b1,
    const bf16* __restrict__ w2, const float* __restrict__ b2,
    float* __restrict__ agg, int E_)
{
    __shared__ bf16 As[64 * KP1];     // 36,864 B: gathered edge inputs (bf16)
    __shared__ bf16 Hs[64 * H_DIM];   // 16,384 B: hidden activations
    __shared__ int  rowS[64];
    __shared__ int  colS[64];
    __shared__ float radS[64];

    const int tid = threadIdx.x;
    const int tileBase = blockIdx.x * 64;

    // per-edge metadata
    if (tid < 64) {
        int e = tileBase + tid;
        int r = 0, c = 0; float rad = 0.0f;
        if (e < E_) {
            r = ei[e];           // edge_index[0][e]
            c = ei[E_ + e];      // edge_index[1][e]
            float dx = coord[r * 3 + 0] - coord[c * 3 + 0];
            float dy = coord[r * 3 + 1] - coord[c * 3 + 1];
            float dz = coord[r * 3 + 2] - coord[c * 3 + 2];
            rad = dx * dx + dy * dy + dz * dz;
        }
        rowS[tid] = r; colS[tid] = c; radS[tid] = rad;
    }
    __syncthreads();

    // build A tile: [64 x 288] = concat(x[row], x[col], radial, edge_attr, 0-pad)
    for (int i = tid; i < 64 * KP1; i += 256) {
        int er = i / KP1, k = i % KP1;
        int e = tileBase + er;
        float v = 0.0f;
        if (e < E_) {
            if (k < 128)       v = x[rowS[er] * C_DIM + k];
            else if (k < 256)  v = x[colS[er] * C_DIM + (k - 128)];
            else if (k == 256) v = radS[er];
            else if (k < 265)  v = ea[e * EA_DIM + (k - 257)];
        }
        As[i] = (bf16)v;
    }
    __syncthreads();

    const int wave = tid >> 5;
    const int lane = tid & 31;
    const int mt = wave & 3;          // 16-edge sub-tile
    const int nh = wave >> 2;         // 64-col half
    const int lm = lane & 15;
    const int kh = lane >> 4;

    // ---- GEMM1: [64x288] @ [288x128] ----
    v8f acc[4] = {};
    #pragma unroll
    for (int kc = 0; kc < KP1 / 32; ++kc) {
        AFrag a;
        const bf16* arow = &As[(mt * 16 + lm) * KP1 + kc * 32 + kh * 16];
        a.h[0] = *(const v8bf*)(arow);
        a.h[1] = *(const v8bf*)(arow + 8);
        #pragma unroll
        for (int nt = 0; nt < 4; ++nt) {
            int n = nh * 64 + nt * 16 + lm;
            v16bf bfr = *(const v16bf*)(&w1[n * KP1 + kc * 32 + kh * 16]);
            acc[nt] = __builtin_amdgcn_wmma_f32_16x16x32_bf16(
                false, a.v, false, bfr, (short)0, acc[nt], false, false);
        }
    }
    // bias + ReLU -> Hs (bf16)
    #pragma unroll
    for (int nt = 0; nt < 4; ++nt) {
        int n = nh * 64 + nt * 16 + lm;
        float bias = b1[n];
        #pragma unroll
        for (int r = 0; r < 8; ++r) {
            float v = acc[nt][r] + bias;
            v = v > 0.0f ? v : 0.0f;
            Hs[(mt * 16 + kh * 8 + r) * H_DIM + n] = (bf16)v;
        }
    }
    __syncthreads();

    // ---- GEMM2: [64x128] @ [128x128] ----
    v8f acc2[4] = {};
    #pragma unroll
    for (int kc = 0; kc < H_DIM / 32; ++kc) {
        AFrag a;
        const bf16* arow = &Hs[(mt * 16 + lm) * H_DIM + kc * 32 + kh * 16];
        a.h[0] = *(const v8bf*)(arow);
        a.h[1] = *(const v8bf*)(arow + 8);
        #pragma unroll
        for (int nt = 0; nt < 4; ++nt) {
            int n = nh * 64 + nt * 16 + lm;
            v16bf bfr = *(const v16bf*)(&w2[n * H_DIM + kc * 32 + kh * 16]);
            acc2[nt] = __builtin_amdgcn_wmma_f32_16x16x32_bf16(
                false, a.v, false, bfr, (short)0, acc2[nt], false, false);
        }
    }
    // bias + ReLU + scatter-add into agg (segment_sum over row)
    #pragma unroll
    for (int nt = 0; nt < 4; ++nt) {
        int n = nh * 64 + nt * 16 + lm;
        float bias = b2[n];
        #pragma unroll
        for (int r = 0; r < 8; ++r) {
            int m = mt * 16 + kh * 8 + r;
            int e = tileBase + m;
            if (e < E_) {
                float v = acc2[nt][r] + bias;
                v = v > 0.0f ? v : 0.0f;
                atomicAdd(&agg[rowS[m] * H_DIM + n], v);
            }
        }
    }
}

// =====================================================================
// node kernel: 64 nodes per block. concat(x, agg) -> (256->128 ReLU)
// -> (128->128 + bias) -> out. WMMA throughout.
// =====================================================================
__global__ __launch_bounds__(256) void node_kernel(
    const float* __restrict__ x, const float* __restrict__ agg,
    const bf16* __restrict__ w1, const float* __restrict__ b1,
    const bf16* __restrict__ w2, const float* __restrict__ b2,
    float* __restrict__ out, int N_)
{
    __shared__ bf16 As[64 * KN1];     // 32,768 B
    __shared__ bf16 Hs[64 * H_DIM];   // 16,384 B

    const int tid = threadIdx.x;
    const int base = blockIdx.x * 64;

    for (int i = tid; i < 64 * KN1; i += 256) {
        int nr = i >> 8, k = i & 255;
        int node = base + nr;
        float v = 0.0f;
        if (node < N_) v = (k < 128) ? x[node * C_DIM + k]
                                     : agg[node * H_DIM + (k - 128)];
        As[i] = (bf16)v;
    }
    __syncthreads();

    const int wave = tid >> 5;
    const int lane = tid & 31;
    const int mt = wave & 3;
    const int nh = wave >> 2;
    const int lm = lane & 15;
    const int kh = lane >> 4;

    // ---- GEMM1: [64x256] @ [256x128] ----
    v8f acc[4] = {};
    #pragma unroll
    for (int kc = 0; kc < KN1 / 32; ++kc) {
        AFrag a;
        const bf16* arow = &As[(mt * 16 + lm) * KN1 + kc * 32 + kh * 16];
        a.h[0] = *(const v8bf*)(arow);
        a.h[1] = *(const v8bf*)(arow + 8);
        #pragma unroll
        for (int nt = 0; nt < 4; ++nt) {
            int n = nh * 64 + nt * 16 + lm;
            v16bf bfr = *(const v16bf*)(&w1[n * KN1 + kc * 32 + kh * 16]);
            acc[nt] = __builtin_amdgcn_wmma_f32_16x16x32_bf16(
                false, a.v, false, bfr, (short)0, acc[nt], false, false);
        }
    }
    #pragma unroll
    for (int nt = 0; nt < 4; ++nt) {
        int n = nh * 64 + nt * 16 + lm;
        float bias = b1[n];
        #pragma unroll
        for (int r = 0; r < 8; ++r) {
            float v = acc[nt][r] + bias;
            v = v > 0.0f ? v : 0.0f;
            Hs[(mt * 16 + kh * 8 + r) * H_DIM + n] = (bf16)v;
        }
    }
    __syncthreads();

    // ---- GEMM2: [64x128] @ [128x128] (no final ReLU) ----
    v8f acc2[4] = {};
    #pragma unroll
    for (int kc = 0; kc < H_DIM / 32; ++kc) {
        AFrag a;
        const bf16* arow = &Hs[(mt * 16 + lm) * H_DIM + kc * 32 + kh * 16];
        a.h[0] = *(const v8bf*)(arow);
        a.h[1] = *(const v8bf*)(arow + 8);
        #pragma unroll
        for (int nt = 0; nt < 4; ++nt) {
            int n = nh * 64 + nt * 16 + lm;
            v16bf bfr = *(const v16bf*)(&w2[n * H_DIM + kc * 32 + kh * 16]);
            acc2[nt] = __builtin_amdgcn_wmma_f32_16x16x32_bf16(
                false, a.v, false, bfr, (short)0, acc2[nt], false, false);
        }
    }
    #pragma unroll
    for (int nt = 0; nt < 4; ++nt) {
        int n = nh * 64 + nt * 16 + lm;
        float bias = b2[n];
        #pragma unroll
        for (int r = 0; r < 8; ++r) {
            int m = mt * 16 + kh * 8 + r;
            int node = base + m;
            if (node < N_)
                out[node * C_DIM + n] = acc2[nt][r] + bias;
        }
    }
}

// =====================================================================
extern "C" void kernel_launch(void* const* d_in, const int* in_sizes, int n_in,
                              void* d_out, int out_size, void* d_ws, size_t ws_size,
                              hipStream_t stream) {
    const float* x     = (const float*)d_in[0];
    const float* coord = (const float*)d_in[1];
    const int*   ei    = (const int*)d_in[2];
    const float* ea    = (const float*)d_in[3];
    const float* w_e1  = (const float*)d_in[4];
    const float* b_e1  = (const float*)d_in[5];
    const float* w_e2  = (const float*)d_in[6];
    const float* b_e2  = (const float*)d_in[7];
    const float* w_n1  = (const float*)d_in[8];
    const float* b_n1  = (const float*)d_in[9];
    const float* w_n2  = (const float*)d_in[10];
    const float* b_n2  = (const float*)d_in[11];
    float* out = (float*)d_out;

    const int N = in_sizes[0] / C_DIM;     // 50000
    const int E = in_sizes[3] / EA_DIM;    // 600000

    // workspace layout (bf16 weights pre-swizzled col-major + f32 agg)
    bf16* w1  = (bf16*)d_ws;               // KP1*128
    bf16* w2  = w1 + KP1 * H_DIM;          // 128*128
    bf16* wn1 = w2 + H_DIM * H_DIM;        // 256*128
    bf16* wn2 = wn1 + KN1 * H_DIM;         // 128*128
    float* agg = (float*)(wn2 + H_DIM * C_DIM);  // N*128 f32 (offset 204,800 B)

    prep_kernel<<<512, 256, 0, stream>>>(w_e1, w_e2, w_n1, w_n2,
                                         w1, w2, wn1, wn2, agg, N * H_DIM);
    edge_kernel<<<(E + 63) / 64, 256, 0, stream>>>(x, coord, ei, ea,
                                                   w1, b_e1, w2, b_e2, agg, E);
    node_kernel<<<(N + 63) / 64, 256, 0, stream>>>(x, agg, wn1, b_n1,
                                                   wn2, b_n2, out, N);
}
